// EGATM_82532091560223
// MI455X (gfx1250) — compile-verified, matrix operand order checked by reading, not run on hardware
//
#include <hip/hip_runtime.h>
#include <hip/hip_bf16.h>

// ---------------------------------------------------------------------------
// EGAT + MLP for MI455X (gfx1250).
// Heavy GEMMs: bf16 WMMA (f32 accum), operands pre-swizzled into the ISA
// 7.12.2 fragment layout so LDS fragment loads are single 32B ds_load_b128
// pairs with no VALU repacking. B tile staged via async global->LDS loads.
// Graph ops use L2-resident float atomics. All launches on `stream`.
// ---------------------------------------------------------------------------

typedef __attribute__((ext_vector_type(16))) __bf16 v16bf;
typedef __attribute__((ext_vector_type(8)))  float  v8f;

#define D_IN  1346
#define D_PAD 1376          // 1346 -> multiple of 32 (K pad, C row stride)
#define K_X   2720          // 2692 -> multiple of 32
#define N_PAD 1408          // 1346 -> multiple of 128 (B col pad)
#define JG_N  (N_PAD / 16)  // 88 column blocks

// fragment swizzle: k within a 32-wide K-tile -> (half, elem) per ISA 7.12.2
__device__ __forceinline__ void frag_he(int kk, int& half, int& e) {
  half = (kk >> 3) & 1;
  e    = (kk & 16) ? (8 + (kk & 7)) : (kk & 7);
}

// ---------------- small helpers ----------------
__device__ __forceinline__ unsigned f2ord(float f) {
  unsigned u = __float_as_uint(f);
  return (u & 0x80000000u) ? ~u : (u | 0x80000000u);
}
__device__ __forceinline__ float ord2f(unsigned o) {
  unsigned u = (o & 0x80000000u) ? (o & 0x7fffffffu) : ~o;
  return __uint_as_float(u);
}

// ---------------- utility kernels ----------------
__global__ void k_zero_f32(float* p, long long n) {
  long long i = (long long)blockIdx.x * blockDim.x + threadIdx.x;
  if (i < n) p[i] = 0.0f;
}

// A-operand: swizzled bf16, index ((rb*KT + kt)*32 + half*16 + (m&15))*16 + e
__global__ void k_cast_A_swz(__bf16* dst, const float* src, int Mpad, int Kpad,
                             int rows_src, int cols_src) {
  long long i = (long long)blockIdx.x * blockDim.x + threadIdx.x;
  long long total = (long long)Mpad * Kpad;
  if (i >= total) return;
  int m = (int)(i / Kpad), k = (int)(i % Kpad);
  float v = (m < rows_src && k < cols_src) ? src[(long long)m * cols_src + k] : 0.f;
  int KT = Kpad >> 5, half, e;
  frag_he(k & 31, half, e);
  size_t idx = (((size_t)(m >> 4) * KT + (k >> 5)) * 32 + half * 16 + (m & 15)) * 16 + e;
  dst[idx] = (__bf16)v;
}

// x = concat(h, h1), swizzled like A, Kpad = K_X
__global__ void k_build_xb_swz(__bf16* dst, const float* h, const float* h1,
                               int N, int Mpad) {
  long long i = (long long)blockIdx.x * blockDim.x + threadIdx.x;
  long long total = (long long)Mpad * K_X;
  if (i >= total) return;
  int m = (int)(i / K_X), k = (int)(i % K_X);
  float v = 0.f;
  if (m < N) {
    if (k < D_IN)          v = h [(long long)m * D_IN + k];
    else if (k < 2 * D_IN) v = h1[(long long)m * D_IN + (k - D_IN)];
  }
  int KT = K_X >> 5, half, e;
  frag_he(k & 31, half, e);
  size_t idx = (((size_t)(m >> 4) * KT + (k >> 5)) * 32 + half * 16 + (m & 15)) * 16 + e;
  dst[idx] = (__bf16)v;
}

// B-operand: swizzled bf16, index ((kt*JG + jg)*32 + half*16 + (n&15))*16 + e
__global__ void k_cast_B_swz(__bf16* dst, const float* src, int Kpad,
                             int rows_src, int cols_src) {
  long long i = (long long)blockIdx.x * blockDim.x + threadIdx.x;
  long long total = (long long)Kpad * N_PAD;
  if (i >= total) return;
  int k = (int)(i / N_PAD), n = (int)(i % N_PAD);
  float v = (k < rows_src && n < cols_src) ? src[(long long)k * cols_src + n] : 0.f;
  int half, e;
  frag_he(k & 31, half, e);
  size_t idx = (((size_t)(k >> 5) * JG_N + (n >> 4)) * 32 + half * 16 + (n & 15)) * 16 + e;
  dst[idx] = (__bf16)v;
}

// ---------------- skinny node projections: f_ni / f_nj (one wave per node) --
__global__ void k_node_lin(const float* __restrict__ h,
                           const float* __restrict__ Wni,
                           const float* __restrict__ Wnj,
                           float* __restrict__ fni, float* __restrict__ fnj, int N) {
  int node = blockIdx.x * (blockDim.x >> 5) + (threadIdx.x >> 5);
  int lane = threadIdx.x & 31;
  float s0 = 0.f, s1 = 0.f, s2 = 0.f, s3 = 0.f;
  if (node < N) {
    const float* hr = h + (long long)node * D_IN;
    for (int k = lane; k < D_IN; k += 32) {
      float x = hr[k];
      s0 += x * Wni[2 * k];     s1 += x * Wni[2 * k + 1];
      s2 += x * Wnj[2 * k];     s3 += x * Wnj[2 * k + 1];
    }
  }
  for (int off = 16; off; off >>= 1) {
    s0 += __shfl_xor(s0, off, 32);  s1 += __shfl_xor(s1, off, 32);
    s2 += __shfl_xor(s2, off, 32);  s3 += __shfl_xor(s3, off, 32);
  }
  if (node < N && lane == 0) {
    fni[2 * node] = s0;  fni[2 * node + 1] = s1;
    fnj[2 * node] = s2;  fnj[2 * node + 1] = s3;
  }
}

// ---------------- edge logits + segment-max (ordered-uint atomicMax) --------
__global__ void k_edge_logit(const float* __restrict__ fni, const float* __restrict__ fnj,
                             const float* __restrict__ ex,  const float* __restrict__ Wfij,
                             const float* __restrict__ bias_e, const float* __restrict__ attn,
                             const int* __restrict__ src, const int* __restrict__ dst,
                             float* __restrict__ logits, unsigned* __restrict__ mbits, int E) {
  int e = blockIdx.x * blockDim.x + threadIdx.x;
  if (e >= E) return;
  int s = src[e], d = dst[e];
  float e0 = ex[2 * e], e1 = ex[2 * e + 1];
  float f0 = fni[2 * s]     + fnj[2 * d]     + e0 * Wfij[0] + e1 * Wfij[2] + bias_e[0];
  float f1 = fni[2 * s + 1] + fnj[2 * d + 1] + e0 * Wfij[1] + e1 * Wfij[3] + bias_e[1];
  f0 = f0 > 0.f ? f0 : 0.01f * f0;           // leaky_relu(0.01)
  f1 = f1 > 0.f ? f1 : 0.01f * f1;
  float lg = attn[0] * f0 + attn[1] * f1;
  logits[e] = lg;
  atomicMax(&mbits[d], f2ord(lg));
}

// ---------------- exp + segment-sum -----------------------------------------
__global__ void k_edge_exp(const float* __restrict__ logits, const int* __restrict__ dst,
                           const unsigned* __restrict__ mbits,
                           float* __restrict__ ez, float* __restrict__ denom, int E) {
  int e = blockIdx.x * blockDim.x + threadIdx.x;
  if (e >= E) return;
  int d = dst[e];
  float z = __expf(logits[e] - ord2f(mbits[d]));
  ez[e] = z;
  atomicAdd(&denom[d], z);
}

// ---------------- h1[dst] += a * h_proj[src]  (one block per edge) ----------
__global__ void k_aggregate(float* __restrict__ h1, const float* __restrict__ hproj,
                            const float* __restrict__ ez, const float* __restrict__ denom,
                            const int* __restrict__ src, const int* __restrict__ dst) {
  int e = blockIdx.x;
  int s = src[e], d = dst[e];
  float a = ez[e] / denom[d];
  const float* hp  = hproj + (long long)s * D_PAD;
  float*       out = h1    + (long long)d * D_IN;
  for (int c = threadIdx.x; c < D_IN; c += blockDim.x)
    atomicAdd(&out[c], a * hp[c]);
}

// ---------------- bf16 WMMA GEMM on pre-swizzled operands -------------------
// Block tile 128(M) x 128(N), 8 waves as 4(M) x 2(N), wave tile 32x64
// => 2x4 v_wmma_f32_16x16x32_bf16 per wave per K-step.
__global__ __launch_bounds__(256) void k_gemm_bf16(
    const __bf16* __restrict__ A, const __bf16* __restrict__ B,
    float* __restrict__ C, const float* __restrict__ bias,
    int M, int Nv, int KT, int ldc, int relu) {
  __shared__ __align__(32) __bf16 As[128 * 32];   // 8 row-block chunks of 512
  __shared__ __align__(32) __bf16 Bs[32 * 128];   // 8 col-block chunks of 512

  const int tid  = threadIdx.x;
  const int wave = tid >> 5, lane = tid & 31;
  const int wm   = wave >> 1;          // 0..3 (M)
  const int wn   = wave & 1;           // 0..1 (N)
  const int rb0  = blockIdx.y * 8;     // first 16-row block
  const int jg0  = blockIdx.x * 8;     // first 16-col block

  v8f acc[2][4] = {};

  for (int kt = 0; kt < KT; ++kt) {
    // stage A: gather 8 x 1KB fragment-native chunks (no guards: A is padded)
#pragma unroll
    for (int c = 0; c < 2; ++c) {
      int u = tid + c * 256;                 // 512 x uint4
      int sb = u >> 6, w = u & 63;
      const uint4* srcp =
          (const uint4*)(A + ((size_t)(rb0 + sb) * KT + kt) * 512) + w;
      *((uint4*)As + u) = *srcp;
    }
    // stage B: one contiguous 8KB block via async global->LDS (CDNA5 path)
    {
      const __bf16* bsrc = B + ((size_t)kt * JG_N + jg0) * 512;
      unsigned lds0 = (unsigned)(size_t)&Bs[0];
#pragma unroll
      for (int c = 0; c < 2; ++c) {
        int u = tid + c * 256;
        unsigned lds = lds0 + (unsigned)u * 16u;
        const __bf16* g = bsrc + (size_t)u * 8;
        asm volatile("global_load_async_to_lds_b128 %0, %1, off"
                     :: "v"(lds), "v"(g) : "memory");
      }
      asm volatile("s_wait_asynccnt 0x0" ::: "memory");
    }
    __syncthreads();

    // fragments: each lane's 16 bf16 are contiguous 32B -> 2x ds_load_b128
    v16bf af[2], bfr[4];
#pragma unroll
    for (int i = 0; i < 2; ++i)
      af[i] = *(const v16bf*)(As + ((wm * 2 + i) * 32 + lane) * 16);
#pragma unroll
    for (int j = 0; j < 4; ++j)
      bfr[j] = *(const v16bf*)(Bs + ((wn * 4 + j) * 32 + lane) * 16);

#pragma unroll
    for (int i = 0; i < 2; ++i)
#pragma unroll
      for (int j = 0; j < 4; ++j)
        acc[i][j] = __builtin_amdgcn_wmma_f32_16x16x32_bf16(
            false, af[i], false, bfr[j], (short)0, acc[i][j], false, false);

    __syncthreads();
  }

  // C/D layout: VGPR r, lane -> row r + 8*half, col l16 (ISA 7.12.2)
  const int half = lane >> 4, l16 = lane & 15;
#pragma unroll
  for (int i = 0; i < 2; ++i) {
#pragma unroll
    for (int j = 0; j < 4; ++j) {
      const int col = (jg0 + wn * 4 + j) * 16 + l16;
      if (col >= Nv) continue;
      const float badd = bias ? bias[col] : 0.0f;
#pragma unroll
      for (int r = 0; r < 8; ++r) {
        const int row = (rb0 + wm * 2 + i) * 16 + half * 8 + r;
        if (row < M) {
          float v = acc[i][j][r] + badd;
          if (relu) v = fmaxf(v, 0.0f);
          C[(size_t)row * ldc + col] = v;
        }
      }
    }
  }
}

// ---------------- fc2: out[n] = b + sum_c relu_y1[n][c] * W_fc2[c] ----------
__global__ void k_fc2(const float* __restrict__ y1, const float* __restrict__ Wfc2,
                      const float* __restrict__ bfc2, float* __restrict__ out, int N) {
  int node = blockIdx.x * (blockDim.x >> 5) + (threadIdx.x >> 5);
  int lane = threadIdx.x & 31;
  float s = 0.f;
  if (node < N) {
    const float* yr = y1 + (long long)node * D_PAD;
    for (int c = lane; c < D_IN; c += 32) s += yr[c] * Wfc2[c];
  }
  for (int off = 16; off; off >>= 1) s += __shfl_xor(s, off, 32);
  if (node < N && lane == 0) out[node] = s + bfc2[0];
}

// ---------------------------------------------------------------------------
extern "C" void kernel_launch(void* const* d_in, const int* in_sizes, int n_in,
                              void* d_out, int out_size, void* d_ws, size_t ws_size,
                              hipStream_t stream) {
  const float* h      = (const float*)d_in[0];
  const int*   src    = (const int*)  d_in[1];
  const int*   dst    = (const int*)  d_in[2];
  const float* ex     = (const float*)d_in[3];
  const float* Wni    = (const float*)d_in[4];
  const float* Wnj    = (const float*)d_in[5];
  const float* Wfij   = (const float*)d_in[6];
  const float* bias_e = (const float*)d_in[7];
  const float* attn   = (const float*)d_in[8];
  const float* Wnode  = (const float*)d_in[9];
  const float* Wfc1   = (const float*)d_in[10];
  const float* bfc1   = (const float*)d_in[11];
  const float* Wfc2   = (const float*)d_in[12];
  const float* bfc2   = (const float*)d_in[13];
  float* out = (float*)d_out;

  const int N    = in_sizes[0] / D_IN;
  const int E    = in_sizes[1];
  const int Mpad = ((N + 127) / 128) * 128;   // 12032

  // ---- scratch layout (256B aligned, with reuse) ----
  char* base = (char*)d_ws;
  size_t off = 0;
  auto alloc = [&](size_t bytes) {
    off = (off + 255) & ~(size_t)255;
    char* p = base + off;
    off += bytes;
    return p;
  };
  __bf16* bufA   = (__bf16*)alloc((size_t)Mpad * K_X * 2);            // h then x, swizzled
  __bf16* bufW   = (__bf16*)alloc((size_t)(K_X / 32) * JG_N * 512 * 2); // W, swizzled
  float*  bufC   = (float*) alloc((size_t)N * D_PAD * 4);             // hproj then y1
  float*  h1     = (float*) alloc((size_t)N * D_IN * 4);
  float*  fni    = (float*) alloc((size_t)N * 2 * 4);
  float*  fnj    = (float*) alloc((size_t)N * 2 * 4);
  float*  logits = (float*) alloc((size_t)E * 4);
  float*  ez     = (float*) alloc((size_t)E * 4);
  unsigned* mbits= (unsigned*)alloc((size_t)N * 4);
  float*  denom  = (float*) alloc((size_t)N * 4);
  (void)ws_size; (void)n_in; (void)out_size;

  const int B256 = 256;
  auto blocks = [](long long n, int b) { return (unsigned)((n + b - 1) / b); };

  // 1) h -> swizzled bf16 A [Mpad x 1376]
  k_cast_A_swz<<<blocks((long long)Mpad * D_PAD, B256), B256, 0, stream>>>(
      bufA, h, Mpad, D_PAD, N, D_IN);
  // 2) W_node -> swizzled bf16 B [1376 x 1408]
  k_cast_B_swz<<<blocks((long long)D_PAD * N_PAD, B256), B256, 0, stream>>>(
      bufW, Wnode, D_PAD, D_IN, D_IN);
  // 3) h_proj = h @ W_node  (WMMA bf16)
  {
    dim3 grid((D_IN + 127) / 128, Mpad / 128);
    k_gemm_bf16<<<grid, 256, 0, stream>>>(bufA, bufW, bufC, nullptr,
                                          N, D_IN, D_PAD / 32, D_PAD, 0);
  }
  // 4) skinny projections
  k_node_lin<<<blocks((long long)N * 32, B256), B256, 0, stream>>>(h, Wni, Wnj, fni, fnj, N);
  // 5) zero accumulators
  k_zero_f32<<<blocks((long long)N * D_IN, B256), B256, 0, stream>>>(h1, (long long)N * D_IN);
  k_zero_f32<<<blocks(N, B256), B256, 0, stream>>>(denom, N);
  k_zero_f32<<<blocks(N, B256), B256, 0, stream>>>((float*)mbits, N);  // ord floor
  // 6) edge logits + segment max
  k_edge_logit<<<blocks(E, B256), B256, 0, stream>>>(fni, fnj, ex, Wfij, bias_e, attn,
                                                     src, dst, logits, mbits, E);
  // 7) exp + segment sum
  k_edge_exp<<<blocks(E, B256), B256, 0, stream>>>(logits, dst, mbits, ez, denom, E);
  // 8) h1[dst] += a * h_proj[src]
  k_aggregate<<<E, 256, 0, stream>>>(h1, bufC, ez, denom, src, dst);
  // 9) x = concat(h, h1) -> swizzled bf16 A [Mpad x 2720]
  k_build_xb_swz<<<blocks((long long)Mpad * K_X, B256), B256, 0, stream>>>(
      bufA, h, h1, N, Mpad);
  // 10) W_fc1 -> swizzled bf16 B [2720 x 1408]
  k_cast_B_swz<<<blocks((long long)K_X * N_PAD, B256), B256, 0, stream>>>(
      bufW, Wfc1, K_X, 2 * D_IN, D_IN);
  // 11) y1 = relu(x @ W_fc1 + b_fc1)  (WMMA bf16, fused bias+relu)
  {
    dim3 grid((D_IN + 127) / 128, Mpad / 128);
    k_gemm_bf16<<<grid, 256, 0, stream>>>(bufA, bufW, bufC, bfc1,
                                          N, D_IN, K_X / 32, D_PAD, 1);
  }
  // 12) out = y1 @ W_fc2 + b_fc2
  k_fc2<<<blocks((long long)N * 32, B256), B256, 0, stream>>>(bufC, Wfc2, bfc2, out, N);
}
